// MindistAwareAttention_6442450944452
// MI455X (gfx1250) — matrix operand order repaired and checked by reading, not compiled, not run on hardware
//
#include <hip/hip_runtime.h>
#include <hip/hip_bf16.h>
#include <math.h>

// ---------------------------------------------------------------------------
// MindistAwareAttention on MI455X (gfx1250, wave32, WMMA f32_16x16x32_f16)
//   k0: bias_table[10] = (emb @ rowsum(Wo)) / sqrt(512)
//   k1: Q = x@Wq^T (f16 row-major), K = x@Wk^T (f16 row-major),
//       V = x@Wv^T stored transposed per batch [d][n] (f16)
//   k2: fused flash attention with distance-level bias; O overwrites Q buffer
//       dist tile staged in LDS via GLOBAL_LOAD_ASYNC_TO_LDS_B128 (ASYNCcnt)
//   k3: out = O @ Wo^T + bo (f32)
// ---------------------------------------------------------------------------

typedef __attribute__((ext_vector_type(16))) _Float16 v16h;
typedef __attribute__((ext_vector_type(8)))  float    v8f;

#define D_MODEL 512
#define SEQ     2048
#define BATCH   8
#define NLEV    10
#define INV_SQRT_D 0.044194173824159216f  // 1/sqrt(512)

union HV { v16h v; uint4 q[2]; };

static __device__ inline v8f wmma16(v16h a, v16h b, v8f c) {
  // (neg_a, A, neg_b, B, c_mod, C, reuse_a, reuse_b)
  return __builtin_amdgcn_wmma_f32_16x16x32_f16(false, a, false, b,
                                                (short)0, c, false, false);
}

// A-fragment (16x32 f16): lane L: row = L%16; K offs (L/16)*8+[0..7], +16..23
static __device__ inline v16h load_a_f16(const _Float16* __restrict__ p) {
  HV t;
  t.q[0] = *(const uint4*)(p);
  t.q[1] = *(const uint4*)(p + 16);
  return t.v;
}
// B-fragment from B^T row-major: lane L: row(N)=L%16; K offs (L/16)*8+[0..7], +16..23
static __device__ inline v16h load_b_f16(const _Float16* __restrict__ p) {
  HV t;
  t.q[0] = *(const uint4*)(p);
  t.q[1] = *(const uint4*)(p + 16);
  return t.v;
}
static __device__ inline v16h cvt16_f32(const float* __restrict__ f) {
  v16h r;
#pragma unroll
  for (int i = 0; i < 16; ++i) r[i] = (_Float16)f[i];
  return r;
}

// ---------------------------------------------------------------------------
// k0: bias table (one workgroup)
// ---------------------------------------------------------------------------
__global__ __launch_bounds__(256) void bias_table_kernel(
    const float* __restrict__ Wo, const float* __restrict__ emb,
    float* __restrict__ btab) {
  __shared__ float wsum[D_MODEL];
  __shared__ float red[256];
  const int tid = threadIdx.x;
  for (int k = tid; k < D_MODEL; k += 256) {
    float s = 0.f;
    const float* row = Wo + (size_t)k * D_MODEL;
    for (int j = 0; j < D_MODEL; ++j) s += row[j];
    wsum[k] = s;
  }
  __syncthreads();
  for (int l = 0; l < NLEV; ++l) {
    float p = 0.f;
    for (int k = tid; k < D_MODEL; k += 256) p += emb[l * D_MODEL + k] * wsum[k];
    red[tid] = p;
    __syncthreads();
    for (int off = 128; off > 0; off >>= 1) {
      if (tid < off) red[tid] += red[tid + off];
      __syncthreads();
    }
    if (tid == 0) btab[l] = red[0] * INV_SQRT_D;
    __syncthreads();
  }
}

// ---------------------------------------------------------------------------
// k1/k3: generic GEMM  C[16384 x 512] = A[16384 x 512] * W^T + bias
//   MODE 0: f16 row-major out      MODE 1: f16 transposed [b][512][2048]
//   MODE 2: f32 row-major + bias   A_F16: A is f16 (else f32, cvt on load)
// ---------------------------------------------------------------------------
template <int MODE, bool A_F16>
__global__ __launch_bounds__(128) void gemm16(
    const void* __restrict__ Ap, const float* __restrict__ W,
    const float* __restrict__ bias, void* __restrict__ Out) {
  const int wave = threadIdx.x >> 5;
  const int lane = threadIdx.x & 31;
  const int lh = lane >> 4;
  const int ln = lane & 15;
  const int m0 = blockIdx.x * 16;
  const int n0 = blockIdx.y * 64 + wave * 16;
  const int arow = m0 + ln;
  const int brow = n0 + ln;

  v8f acc = {0.f, 0.f, 0.f, 0.f, 0.f, 0.f, 0.f, 0.f};

  for (int kc = 0; kc < D_MODEL; kc += 32) {
    v16h a;
    if constexpr (A_F16) {
      const _Float16* A = (const _Float16*)Ap;
      a = load_a_f16(A + (size_t)arow * D_MODEL + kc + lh * 8);
    } else {
      const float* A = (const float*)Ap;
      const float* p = A + (size_t)arow * D_MODEL + kc + lh * 8;
      float af[16];
      *(float4*)(af + 0)  = *(const float4*)(p + 0);
      *(float4*)(af + 4)  = *(const float4*)(p + 4);
      *(float4*)(af + 8)  = *(const float4*)(p + 16);
      *(float4*)(af + 12) = *(const float4*)(p + 20);
      a = cvt16_f32(af);
    }
    const float* wp = W + (size_t)brow * D_MODEL + kc + lh * 8;
    float bf[16];
    *(float4*)(bf + 0)  = *(const float4*)(wp + 0);
    *(float4*)(bf + 4)  = *(const float4*)(wp + 4);
    *(float4*)(bf + 8)  = *(const float4*)(wp + 16);
    *(float4*)(bf + 12) = *(const float4*)(wp + 20);
    v16h b = cvt16_f32(bf);
    acc = wmma16(a, b, acc);
  }

  const float bv = bias[brow];

  if constexpr (MODE == 0) {
    _Float16* O = (_Float16*)Out;
#pragma unroll
    for (int r = 0; r < 8; ++r)
      O[(size_t)(m0 + r + 8 * lh) * D_MODEL + brow] = (_Float16)(acc[r] + bv);
  } else if constexpr (MODE == 1) {
    _Float16* O = (_Float16*)Out;
    const int bidx = m0 / SEQ;
    const int mloc = (m0 % SEQ) + 8 * lh;
    union { _Float16 h[8]; uint4 q; } hv;
#pragma unroll
    for (int r = 0; r < 8; ++r) hv.h[r] = (_Float16)(acc[r] + bv);
    *(uint4*)(O + (size_t)bidx * D_MODEL * SEQ + (size_t)brow * SEQ + mloc) = hv.q;
  } else {
    float* O = (float*)Out;
#pragma unroll
    for (int r = 0; r < 8; ++r)
      O[(size_t)(m0 + r + 8 * lh) * D_MODEL + brow] = acc[r] + bv;
  }
}

// ---------------------------------------------------------------------------
// k2: fused flash attention with distance-level bias.
// Grid (SEQ/16, BATCH); block = 128 thr = 4 waves, 16 query rows per block.
// Wave w contracts dk slice [128w,128w+128) for scores, owns dv slice
// [128w,128w+128) of O. Partial scores summed through LDS. The 16x32 f32
// dist tile is DMA'd to LDS once per chunk (async, ASYNCcnt) instead of
// being loaded 4x redundantly by every wave.
// ---------------------------------------------------------------------------
__global__ __launch_bounds__(128) void attn_kernel(
    _Float16* __restrict__ Qb,            // in: Q ; out: attn@V (reused)
    const _Float16* __restrict__ Kb,      // [b][n][d] f16
    const _Float16* __restrict__ Vt,      // [b][d][n] f16
    const float* __restrict__ dist,       // [b][n][n] f32
    const float* __restrict__ btab,       // [10]
    const float* __restrict__ thr_p) {
  __shared__ float Sps[4 * 2 * 8 * 32];           // 8 KB partial scores
  __shared__ __align__(16) _Float16 Pb[16 * 32];  // 1 KB P tile (row-major)
  __shared__ __align__(16) float Dt[16 * 32];     // 2 KB dist tile
  __shared__ float bt[16];

  const int wave = threadIdx.x >> 5;
  const int lane = threadIdx.x & 31;
  const int lh = lane >> 4;
  const int ln = lane & 15;
  const int bidx = blockIdx.y;
  const int i0 = blockIdx.x * 16;
  const int kbase = wave * 128;
  const float thr = *thr_p;

  if (threadIdx.x < NLEV) bt[threadIdx.x] = btab[threadIdx.x];

  const size_t qoff = (size_t)bidx * SEQ * D_MODEL;
  const size_t doff = (size_t)bidx * SEQ * SEQ;
  const size_t voff = (size_t)bidx * D_MODEL * SEQ;

  // dist tile staging map: thread -> (row, 4-float column group), 16B each
  const int drow = threadIdx.x >> 3;
  const int dcol = (threadIdx.x & 7) * 4;
  const unsigned dt_lds = (unsigned)(uintptr_t)(&Dt[drow * 32 + dcol]);
  const float* dgbase = dist + doff + (size_t)(i0 + drow) * SEQ + dcol;

  // Resident Q A-fragments for this wave's dk slice (4 x 32 = 128)
  v16h qf[4];
#pragma unroll
  for (int f = 0; f < 4; ++f)
    qf[f] = load_a_f16(Qb + qoff + (size_t)(i0 + ln) * D_MODEL + kbase + 32 * f + lh * 8);

  float m[8], l[8];
  v8f acc[8];
#pragma unroll
  for (int r = 0; r < 8; ++r) { m[r] = -3.0e38f; l[r] = 0.f; }
#pragma unroll
  for (int t = 0; t < 8; ++t) acc[t] = (v8f){0.f,0.f,0.f,0.f,0.f,0.f,0.f,0.f};

  for (int j0 = 0; j0 < SEQ; j0 += 32) {
    // ---- kick off async DMA of the dist tile into LDS (overlaps WMMAs) ----
    {
      const float* gp = dgbase + j0;
      asm volatile("global_load_async_to_lds_b128 %0, %1, off"
                   :: "v"(dt_lds), "v"(gp)
                   : "memory");
    }

    // ---- partial scores for 2 key tiles over this wave's dk slice ----
    v8f s[2];
    s[0] = (v8f){0.f,0.f,0.f,0.f,0.f,0.f,0.f,0.f};
    s[1] = (v8f){0.f,0.f,0.f,0.f,0.f,0.f,0.f,0.f};
#pragma unroll
    for (int t = 0; t < 2; ++t) {
      v16h kf[4];
#pragma unroll
      for (int f = 0; f < 4; ++f)
        kf[f] = load_b_f16(Kb + qoff + (size_t)(j0 + t * 16 + ln) * D_MODEL +
                           kbase + 32 * f + lh * 8);
#pragma unroll
      for (int f = 0; f < 4; ++f) s[t] = wmma16(qf[f], kf[f], s[t]);
    }
    if (j0 + 32 < SEQ) {  // prefetch next K tile -> global_prefetch_b8
      __builtin_prefetch(Kb + qoff + (size_t)(j0 + 32 + ln) * D_MODEL + kbase, 0, 1);
      __builtin_prefetch(Vt + voff + (size_t)(kbase + ln) * SEQ + j0 + 32, 0, 1);
    }
#pragma unroll
    for (int t = 0; t < 2; ++t)
#pragma unroll
      for (int r = 0; r < 8; ++r)
        Sps[((wave * 2 + t) * 8 + r) * 32 + lane] = s[t][r];

    // our async write must be complete before the barrier makes it visible
    asm volatile("s_wait_asynccnt 0x0" ::: "memory");
    __syncthreads();

    // ---- reduce partials, add distance bias, online softmax ----
    float fv[2][8];
#pragma unroll
    for (int t = 0; t < 2; ++t) {
#pragma unroll
      for (int r = 0; r < 8; ++r) {
        const int idx = (t * 8 + r) * 32 + lane;
        float sum = Sps[idx] + Sps[idx + 512] + Sps[idx + 1024] + Sps[idx + 1536];
        const float dv = Dt[(r + 8 * lh) * 32 + t * 16 + ln];
        int lev = (int)(dv / thr);
        lev = lev < 0 ? 0 : (lev > NLEV - 1 ? NLEV - 1 : lev);
        fv[t][r] = sum * INV_SQRT_D + bt[lev];
      }
    }
#pragma unroll
    for (int r = 0; r < 8; ++r) {
      float mx = fmaxf(fv[0][r], fv[1][r]);
      mx = fmaxf(mx, __shfl_xor(mx, 1));
      mx = fmaxf(mx, __shfl_xor(mx, 2));
      mx = fmaxf(mx, __shfl_xor(mx, 4));
      mx = fmaxf(mx, __shfl_xor(mx, 8));
      const float mnew = fmaxf(m[r], mx);
      const float alpha = __expf(m[r] - mnew);
      const float p0 = __expf(fv[0][r] - mnew);
      const float p1 = __expf(fv[1][r] - mnew);
      fv[0][r] = p0; fv[1][r] = p1;
      float rs = p0 + p1;
      rs += __shfl_xor(rs, 1);
      rs += __shfl_xor(rs, 2);
      rs += __shfl_xor(rs, 4);
      rs += __shfl_xor(rs, 8);
      l[r] = l[r] * alpha + rs;
      m[r] = mnew;
#pragma unroll
      for (int dt = 0; dt < 8; ++dt) acc[dt][r] *= alpha;
    }
    // ---- publish P (identical across waves; wave 0 writes) ----
    if (wave == 0) {
#pragma unroll
      for (int t = 0; t < 2; ++t)
#pragma unroll
        for (int r = 0; r < 8; ++r)
          Pb[(r + 8 * lh) * 32 + t * 16 + ln] = (_Float16)fv[t][r];
    }
    __syncthreads();

    // ---- O += P x V on this wave's dv slice ----
    v16h pa = load_a_f16(Pb + ln * 32 + lh * 8);
#pragma unroll
    for (int dt = 0; dt < 8; ++dt) {
      v16h vb = load_b_f16(Vt + voff + (size_t)(kbase + dt * 16 + ln) * SEQ +
                           j0 + lh * 8);
      acc[dt] = wmma16(pa, vb, acc[dt]);
    }
  }

  // ---- finalize: O /= l, overwrite Q buffer (rows are block-private) ----
#pragma unroll
  for (int dt = 0; dt < 8; ++dt) {
#pragma unroll
    for (int r = 0; r < 8; ++r) {
      const float o = acc[dt][r] / l[r];
      Qb[qoff + (size_t)(i0 + r + 8 * lh) * D_MODEL + kbase + dt * 16 + ln] =
          (_Float16)o;
    }
  }
}

// ---------------------------------------------------------------------------
extern "C" void kernel_launch(void* const* d_in, const int* in_sizes, int n_in,
                              void* d_out, int out_size, void* d_ws, size_t ws_size,
                              hipStream_t stream) {
  const float* x    = (const float*)d_in[0];
  const float* dist = (const float*)d_in[1];
  const float* Wq   = (const float*)d_in[2];
  const float* bq   = (const float*)d_in[3];
  const float* Wk   = (const float*)d_in[4];
  const float* bk   = (const float*)d_in[5];
  const float* Wv   = (const float*)d_in[6];
  const float* bv   = (const float*)d_in[7];
  const float* Wo   = (const float*)d_in[8];
  const float* bo   = (const float*)d_in[9];
  const float* emb  = (const float*)d_in[10];
  const float* thr  = (const float*)d_in[11];
  float* out = (float*)d_out;

  const size_t qkv = (size_t)BATCH * SEQ * D_MODEL;  // 8,388,608 elems
  _Float16* Qb = (_Float16*)d_ws;       // 16 MB ; becomes attn@V output
  _Float16* Kb = Qb + qkv;              // 16 MB
  _Float16* Vt = Kb + qkv;              // 16 MB, per-batch transposed [d][n]
  float* btab  = (float*)(Vt + qkv);    // 64 B

  bias_table_kernel<<<1, 256, 0, stream>>>(Wo, emb, btab);

  dim3 gg(BATCH * SEQ / 16, D_MODEL / 64);  // 1024 x 8 blocks
  gemm16<0, false><<<gg, 128, 0, stream>>>(x, Wq, bq, Qb);
  gemm16<0, false><<<gg, 128, 0, stream>>>(x, Wk, bk, Kb);
  gemm16<1, false><<<gg, 128, 0, stream>>>(x, Wv, bv, Vt);

  attn_kernel<<<dim3(SEQ / 16, BATCH), 128, 0, stream>>>(Qb, Kb, Vt, dist, btab, thr);

  gemm16<2, true><<<gg, 128, 0, stream>>>(Qb, Wo, bo, out);
}
